// AxialAttention_47236050322079
// MI455X (gfx1250) — compile-verified, matrix operand order checked by reading, not run on hardware
//
#include <hip/hip_runtime.h>

typedef __attribute__((ext_vector_type(16))) __bf16 v16bf;
typedef __attribute__((ext_vector_type(8)))  float  v8f;

#define HIDDEN   1024
#define HEADS    16
#define HEAD_DIM 64
#define CTX      256
#define IMG      32
#define NTOK     1280            // CTX + IMG*IMG
#define BATCH    16
#define NROWS    (BATCH * NTOK)  // 20480
#define QKV_N    (3 * HIDDEN)    // 3072

union Frag {
    unsigned int   u32[8];
    unsigned short u16[16];
    uint4          q[2];
    v16bf          v;
};

__device__ __forceinline__ unsigned short f2bf(float f) {
    unsigned int u = __builtin_bit_cast(unsigned int, f);
    u += 0x7FFFu + ((u >> 16) & 1u);           // round-to-nearest-even
    return (unsigned short)(u >> 16);
}

__global__ void cvt_f32_bf16(const float* __restrict__ in,
                             unsigned short* __restrict__ out, int n) {
    int i = blockIdx.x * blockDim.x + threadIdx.x;
    int stride = gridDim.x * blockDim.x;
    for (; i < n; i += stride) out[i] = f2bf(in[i]);
}

// C = A(MxK,bf16) * B(KxN,bf16); 4 waves, 64x64 tile, K-stage 64.
// Software-pipelined: global loads of stage s+1 overlap WMMAs of stage s;
// global_prefetch warms GL2 two stages ahead. B staged transposed in LDS so
// all fragment gathers are 2x ds_load_b128.
__global__ __launch_bounds__(128)
void gemm_bf16(const unsigned short* __restrict__ A,
               const unsigned short* __restrict__ B,
               void* __restrict__ C,
               const float* __restrict__ bias,
               int N, int K, int scaleCols, int outBf16)
{
    __shared__ unsigned short lA[64][64];    // row-major M x Kstage (8 KB)
    __shared__ unsigned short lBT[64][72];   // transposed [n][k], 144B rows (9 KB)

    const int tid  = threadIdx.x;
    const int wave = tid >> 5;
    const int lane = tid & 31;
    const int m    = lane & 15;
    const int half = lane >> 4;
    const long long rowBlock = (long long)blockIdx.y * 64;
    const int colBlock = blockIdx.x * 64;

    v8f acc[4];
    #pragma unroll
    for (int j = 0; j < 4; ++j)
        #pragma unroll
        for (int e = 0; e < 8; ++e) acc[j][e] = 0.f;

    const int arow = tid >> 1, acol = (tid & 1) << 5;   // A-tile loader mapping
    const int n0 = lane << 1;       // transpose-writer column pair (0..62)
    const int kq = wave << 4;       // 16-row k-group 0/16/32/48

    const unsigned short* aPtr = A + (rowBlock + arow) * K + acol;
    const unsigned short* bPtr = B + colBlock + n0;

    uint4 ra[4];
    unsigned int rb[16];
    auto gload = [&](int k0) {
        const uint4* srcA = (const uint4*)(aPtr + k0);
        ra[0] = srcA[0]; ra[1] = srcA[1]; ra[2] = srcA[2]; ra[3] = srcA[3];
        #pragma unroll
        for (int kk = 0; kk < 16; ++kk)
            rb[kk] = *(const unsigned int*)(bPtr + (long long)(k0 + kq + kk) * N);
    };

    gload(0);
    const int nst = K >> 6;
    for (int s = 0; s < nst; ++s) {
        if (s) __syncthreads();          // previous stage finished reading LDS
        {   // regs -> LDS (A row-major, B transposed via register pack)
            uint4* dst = (uint4*)&lA[arow][acol];
            dst[0] = ra[0]; dst[1] = ra[1]; dst[2] = ra[2]; dst[3] = ra[3];
            union { unsigned short sh[16]; uint4 q[2]; } cA, cB;
            #pragma unroll
            for (int kk = 0; kk < 16; ++kk) {
                cA.sh[kk] = (unsigned short)rb[kk];
                cB.sh[kk] = (unsigned short)(rb[kk] >> 16);
            }
            uint4* d0 = (uint4*)&lBT[n0][kq];
            d0[0] = cA.q[0]; d0[1] = cA.q[1];
            uint4* d1 = (uint4*)&lBT[n0 + 1][kq];
            d1[0] = cB.q[0]; d1[1] = cB.q[1];
        }
        __syncthreads();
        if (s + 1 < nst) gload((s + 1) << 6);   // overlap with WMMAs below
        if (s + 2 < nst) {
            __builtin_prefetch(aPtr + ((s + 2) << 6), 0, 1);
            __builtin_prefetch(bPtr + (long long)(((s + 2) << 6) + kq) * N, 0, 1);
        }
        #pragma unroll
        for (int ks = 0; ks < 2; ++ks) {
            Frag a;   // ISA bf16 A layout: lanes 0-15 K 0-7/16-23, 16-31 K 8-15/24-31
            a.q[0] = *(const uint4*)&lA[wave * 16 + m][ks * 32 + half * 8];
            a.q[1] = *(const uint4*)&lA[wave * 16 + m][ks * 32 + 16 + half * 8];
            #pragma unroll
            for (int j = 0; j < 4; ++j) {
                Frag b;
                b.q[0] = *(const uint4*)&lBT[j * 16 + m][ks * 32 + half * 8];
                b.q[1] = *(const uint4*)&lBT[j * 16 + m][ks * 32 + 16 + half * 8];
                acc[j] = __builtin_amdgcn_wmma_f32_16x16x32_bf16(
                    false, a.v, false, b.v, (short)0, acc[j], false, false);
            }
        }
    }
    #pragma unroll
    for (int j = 0; j < 4; ++j) {
        #pragma unroll
        for (int i = 0; i < 8; ++i) {
            long long row = rowBlock + wave * 16 + i + half * 8;
            int col = colBlock + j * 16 + m;
            float v = acc[j][i];
            if (bias) v += bias[col];
            if (col < scaleCols) v *= 0.125f;   // HEAD_DIM^-0.5, exact
            if (outBf16) ((unsigned short*)C)[row * (long long)N + col] = f2bf(v);
            else         ((float*)C)[row * (long long)N + col] = v;
        }
    }
}

// One block (2 waves, 64 threads) per 32-query-row group.
// g<8: ctx group g, keys = 256 ctx tokens. g>=8: image row a=g-8, keys = 256+32.
__global__ __launch_bounds__(64)
void axial_attn(const unsigned short* __restrict__ qkv,
                unsigned short* __restrict__ outAttn)
{
    __shared__ float          S[32][288];    // scores/probs fp32 (36 KB)
    __shared__ unsigned short Q[32][64];     // 4 KB
    __shared__ unsigned short Kt[16][72];    // K tile row-major (per-row gathers)
    __shared__ unsigned short VT[64][40];    // V tile transposed [n][k] (5 KB)

    const int tid  = threadIdx.x;
    const int wave = tid >> 5;
    const int lane = tid & 31;
    const int m    = lane & 15;
    const int half = lane >> 4;

    const int g = blockIdx.x % 40;
    const int h = (blockIdx.x / 40) % HEADS;
    const int b = blockIdx.x / (40 * HEADS);
    const int base  = b * NTOK;
    const int isImg = (g >= 8);
    const int qBase = isImg ? (base + CTX + (g - 8) * IMG) : (base + g * 32);
    const int nkeys = isImg ? (CTX + IMG) : CTX;
    auto ktok = [&](int j) { return (j < CTX) ? (base + j) : (qBase + (j - CTX)); };

    {   // stage Q (32x64): 2 threads/row, 64B each
        int r = tid >> 1, c = (tid & 1) << 5;
        const uint4* src = (const uint4*)(qkv + (long long)(qBase + r) * QKV_N
                                          + h * HEAD_DIM + c);
        uint4* dst = (uint4*)&Q[r][c];
        dst[0] = src[0]; dst[1] = src[1]; dst[2] = src[2]; dst[3] = src[3];
    }
    __syncthreads();

    Frag aq[2];   // q fragments K=0..31, 32..63 (reused across all key tiles)
    #pragma unroll
    for (int ks = 0; ks < 2; ++ks) {
        aq[ks].q[0] = *(const uint4*)&Q[wave * 16 + m][ks * 32 + half * 8];
        aq[ks].q[1] = *(const uint4*)&Q[wave * 16 + m][ks * 32 + 16 + half * 8];
    }

    // ---- phase 1: S = q @ k^T ----
    const int njt = nkeys >> 4;
    const int kr = tid >> 2, kc = (tid & 3) << 4;    // K-tile loader mapping
    for (int j = 0; j < njt; ++j) {
        {   // stage K tile (16x64), row-major
            const uint4* src = (const uint4*)(qkv + (long long)ktok(j * 16 + kr) * QKV_N
                                              + HIDDEN + h * HEAD_DIM + kc);
            uint4* dst = (uint4*)&Kt[kr][kc];
            dst[0] = src[0]; dst[1] = src[1];
        }
        if (j + 1 < njt)
            __builtin_prefetch(qkv + (long long)ktok((j + 1) * 16 + kr) * QKV_N
                               + HIDDEN + h * HEAD_DIM + kc, 0, 1);
        __syncthreads();
        v8f acc;
        #pragma unroll
        for (int e = 0; e < 8; ++e) acc[e] = 0.f;
        #pragma unroll
        for (int ks = 0; ks < 2; ++ks) {
            Frag bk;   // B(k,n) = K[n][k]: lane n reads its own K row, contiguous
            bk.q[0] = *(const uint4*)&Kt[m][ks * 32 + half * 8];
            bk.q[1] = *(const uint4*)&Kt[m][ks * 32 + 16 + half * 8];
            acc = __builtin_amdgcn_wmma_f32_16x16x32_bf16(
                false, aq[ks].v, false, bk.v, (short)0, acc, false, false);
        }
        #pragma unroll
        for (int i = 0; i < 8; ++i)
            S[wave * 16 + i + half * 8][j * 16 + m] = acc[i];
        __syncthreads();
    }

    // ---- phase 2: softmax rows (fp32, float4-vectorized) ----
    if (tid < 32) {
        const int n4 = nkeys >> 2;
        float4* row = (float4*)&S[tid][0];
        float mx = -1e30f;
        for (int c = 0; c < n4; ++c) {
            float4 v = row[c];
            mx = fmaxf(mx, fmaxf(fmaxf(v.x, v.y), fmaxf(v.z, v.w)));
        }
        float sum = 0.f;
        for (int c = 0; c < n4; ++c) {
            float4 v = row[c];
            v.x = __expf(v.x - mx); v.y = __expf(v.y - mx);
            v.z = __expf(v.z - mx); v.w = __expf(v.w - mx);
            sum += (v.x + v.y) + (v.z + v.w);
            row[c] = v;
        }
        float inv = 1.f / sum;
        for (int c = 0; c < n4; ++c) {
            float4 v = row[c];
            v.x *= inv; v.y *= inv; v.z *= inv; v.w *= inv;
            row[c] = v;
        }
    }
    __syncthreads();

    // ---- phase 3: O = P @ V ----
    v8f accO[4];
    #pragma unroll
    for (int j = 0; j < 4; ++j)
        #pragma unroll
        for (int e = 0; e < 8; ++e) accO[j][e] = 0.f;

    const int n0 = lane << 1;       // transpose-writer column pair
    const int kq = wave << 4;       // k-group 0/16
    const int nks = nkeys >> 5;     // 8 or 9 K-steps of 32
    for (int kk = 0; kk < nks; ++kk) {
        {   // stage V tile (32x64) transposed: coalesced b32 reads, b128 writes
            union { unsigned short sh[16]; uint4 q[2]; } cA, cB;
            #pragma unroll
            for (int r = 0; r < 16; ++r) {
                unsigned int w = *(const unsigned int*)(qkv
                    + (long long)ktok(kk * 32 + kq + r) * QKV_N
                    + 2 * HIDDEN + h * HEAD_DIM + n0);
                cA.sh[r] = (unsigned short)w;
                cB.sh[r] = (unsigned short)(w >> 16);
            }
            uint4* d0 = (uint4*)&VT[n0][kq];
            d0[0] = cA.q[0]; d0[1] = cA.q[1];
            uint4* d1 = (uint4*)&VT[n0 + 1][kq];
            d1[0] = cB.q[0]; d1[1] = cB.q[1];
        }
        if (kk + 1 < nks)
            __builtin_prefetch(qkv + (long long)ktok((kk + 1) * 32 + kq) * QKV_N
                               + 2 * HIDDEN + h * HEAD_DIM + n0, 0, 1);
        __syncthreads();
        Frag ap;   // P fragment: 4x float4 from S row + register bf16 convert
        {
            const float* srow = &S[wave * 16 + m][kk * 32];
            float4 f0 = *(const float4*)(srow + half * 8);
            float4 f1 = *(const float4*)(srow + half * 8 + 4);
            float4 f2 = *(const float4*)(srow + 16 + half * 8);
            float4 f3 = *(const float4*)(srow + 16 + half * 8 + 4);
            float tf[16] = { f0.x, f0.y, f0.z, f0.w,  f1.x, f1.y, f1.z, f1.w,
                             f2.x, f2.y, f2.z, f2.w,  f3.x, f3.y, f3.z, f3.w };
            #pragma unroll
            for (int e = 0; e < 16; ++e) ap.u16[e] = f2bf(tf[e]);
        }
        #pragma unroll
        for (int j = 0; j < 4; ++j) {
            Frag bv;
            bv.q[0] = *(const uint4*)&VT[j * 16 + m][half * 8];
            bv.q[1] = *(const uint4*)&VT[j * 16 + m][16 + half * 8];
            accO[j] = __builtin_amdgcn_wmma_f32_16x16x32_bf16(
                false, ap.v, false, bv.v, (short)0, accO[j], false, false);
        }
        __syncthreads();
    }

    #pragma unroll
    for (int j = 0; j < 4; ++j)
        #pragma unroll
        for (int i = 0; i < 8; ++i) {
            int ri  = wave * 16 + i + half * 8;
            int tok = qBase + ri;
            int col = h * HEAD_DIM + j * 16 + m;
            outAttn[(long long)tok * HIDDEN + col] = f2bf(accO[j][i]);
        }
}

extern "C" void kernel_launch(void* const* d_in, const int* in_sizes, int n_in,
                              void* d_out, int out_size, void* d_ws, size_t ws_size,
                              hipStream_t stream) {
    (void)in_sizes; (void)n_in; (void)out_size; (void)ws_size;
    const float* x    = (const float*)d_in[0];
    const float* Wqkv = (const float*)d_in[1];
    const float* Wout = (const float*)d_in[2];
    const float* bout = (const float*)d_in[3];
    float* out = (float*)d_out;

    size_t off = 0;
    auto wsAlloc = [&](size_t bytes) {
        void* p = (char*)d_ws + off;
        off += (bytes + 255) & ~(size_t)255;
        return p;
    };
    unsigned short* xb   = (unsigned short*)wsAlloc((size_t)NROWS * HIDDEN * 2);
    unsigned short* wqb  = (unsigned short*)wsAlloc((size_t)HIDDEN * QKV_N * 2);
    unsigned short* wob  = (unsigned short*)wsAlloc((size_t)HIDDEN * HIDDEN * 2);
    unsigned short* qkvb = (unsigned short*)wsAlloc((size_t)NROWS * QKV_N * 2);
    unsigned short* atb  = (unsigned short*)wsAlloc((size_t)NROWS * HIDDEN * 2);

    cvt_f32_bf16<<<4096, 256, 0, stream>>>(x, xb, NROWS * HIDDEN);
    cvt_f32_bf16<<<2048, 256, 0, stream>>>(Wqkv, wqb, HIDDEN * QKV_N);
    cvt_f32_bf16<<<1024, 256, 0, stream>>>(Wout, wob, HIDDEN * HIDDEN);

    // qkv = x @ W_qkv, q columns (< HIDDEN) pre-scaled by HEAD_DIM^-0.5
    gemm_bf16<<<dim3(QKV_N / 64, NROWS / 64), 128, 0, stream>>>(
        xb, wqb, qkvb, nullptr, QKV_N, HIDDEN, /*scaleCols=*/HIDDEN, /*outBf16=*/1);

    // fused attention (ctx + axial rows), softmax fp32
    axial_attn<<<BATCH * HEADS * 40, 64, 0, stream>>>(qkvb, atb);

    // out = attn @ W_out + b_out (fp32 store)
    gemm_bf16<<<dim3(HIDDEN / 64, NROWS / 64), 128, 0, stream>>>(
        atb, wob, out, bout, HIDDEN, HIDDEN, /*scaleCols=*/0, /*outBf16=*/0);
}
